// _GraphChebConv_80307298501259
// MI455X (gfx1250) — matrix operand order repaired and checked by reading, not compiled, not run on hardware
//
#include <hip/hip_runtime.h>

// Problem constants (from reference)
#define BQ    16384
#define NN    17
#define CINC  128
#define COUTC 128
#define KORD  3

// Tiling
#define BB       16                    // batches per workgroup
#define HROWS    (BB * NN)             // 272 rows per workgroup = 17 tiles of 16
#define NTHREADS 544                   // 17 waves, one 16-row tile per wave
#define HSTRIDE  136                   // bf16 elems per H row (128 + 8 pad, 272B)
#define WPACK_DW (KORD * 4 * 8 * 256)  // 24576 dwords of prepacked bf16 weights

#define LDS_W_BYTES (WPACK_DW * 4)              // 98304
#define LDS_H_BYTES (3 * HROWS * HSTRIDE * 2)   // 221952
#define LDS_T_OFF   (LDS_W_BYTES + LDS_H_BYTES)
#define LDS_TOTAL   (LDS_T_OFF + 2 * NN * NN * 4) // 322568 <= 327680

typedef __attribute__((ext_vector_type(16))) __bf16 v16bf;
typedef __attribute__((ext_vector_type(8)))  float  v8f;

struct B32x2 { uint4 lo, hi; };

static __device__ __forceinline__ unsigned short f2bf(float f) {
  unsigned int u = __float_as_uint(f);
  u += 0x7fffu + ((u >> 16) & 1u);   // round-to-nearest-even
  return (unsigned short)(u >> 16);
}
static __device__ __forceinline__ float bf2f(unsigned short h) {
  return __uint_as_float(((unsigned int)h) << 16);
}

// ---------------------------------------------------------------------------
// Setup: build T1 = L, T2 = 2L^2 - I on-device; prepack W into bf16 B-fragments
// B-frag layout per (k,kc,ct): 256 dwords; half h at +128 dw; lane at +4 dw;
// dword j holds bf16 pair K = h*16 + hi*8 + 2j (+ kc*32), column = ct*16 + (lane&15)
// ---------------------------------------------------------------------------
__global__ __launch_bounds__(256) void cheb_setup(
    const float* __restrict__ adj, const float* __restrict__ weight,
    float* __restrict__ Tmat, unsigned int* __restrict__ wpack)
{
  __shared__ float dinv[NN];
  __shared__ float Lsh[NN * NN];
  const int tid = threadIdx.x;

  if (tid < NN) {
    float s = 0.f;
    for (int m = 0; m < NN; ++m) s += adj[tid * NN + m];
    dinv[tid] = rsqrtf(s);
  }
  __syncthreads();
  if (tid < NN * NN) {
    int n = tid / NN, m = tid % NN;
    float L = ((n == m) ? 1.f : 0.f) - dinv[n] * adj[n * NN + m] * dinv[m];
    Lsh[tid] = L;
    Tmat[tid] = L;                                   // T1
  }
  __syncthreads();
  if (tid < NN * NN) {
    int n = tid / NN, m = tid % NN;
    float acc = 0.f;
    for (int j = 0; j < NN; ++j) acc += Lsh[n * NN + j] * Lsh[j * NN + m];
    Tmat[NN * NN + tid] = 2.f * acc - ((n == m) ? 1.f : 0.f);  // T2
  }

  for (int i = tid; i < WPACK_DW; i += 256) {
    int j    =  i        & 3;
    int lane = (i >> 2)  & 31;
    int h    = (i >> 7)  & 1;
    int ct   = (i >> 8)  & 7;
    int kc   = (i >> 11) & 3;
    int k    =  i >> 13;
    int ncol = lane & 15;
    int hi   = lane >> 4;
    int K0   = kc * 32 + h * 16 + hi * 8 + 2 * j;
    int col  = ct * 16 + ncol;
    unsigned short lo16 = f2bf(weight[(k * CINC + K0    ) * COUTC + col]);
    unsigned short hi16 = f2bf(weight[(k * CINC + K0 + 1) * COUTC + col]);
    wpack[i] = (unsigned int)lo16 | ((unsigned int)hi16 << 16);
  }
}

// ---------------------------------------------------------------------------
// Fused main kernel: stage x (bf16) -> aggregate T1/T2 hops -> 3-way GEMM via
// v_wmma_f32_16x16x32_bf16 -> bias+relu -> f32 out.
// 1 block = 16 batches = 272 rows = 17 row-tiles; 17 waves, one tile per wave.
// ---------------------------------------------------------------------------
__global__ __launch_bounds__(NTHREADS) void cheb_main(
    const float* __restrict__ x, const float* __restrict__ bias,
    const float* __restrict__ Tmat, const unsigned int* __restrict__ wpack,
    float* __restrict__ out)
{
  extern __shared__ char smem[];
  unsigned int*   wl = (unsigned int*)smem;
  unsigned short* hl = (unsigned short*)(smem + LDS_W_BYTES);
  float*          tl = (float*)(smem + LDS_T_OFF);

  const int tid = threadIdx.x;
  const long long rowbase = (long long)blockIdx.x * HROWS;

  // Stage prepacked weights + T matrices into LDS
  for (int i = tid; i < WPACK_DW; i += NTHREADS) wl[i] = wpack[i];
  for (int i = tid; i < 2 * NN * NN; i += NTHREADS) tl[i] = Tmat[i];

  // Stage X -> H plane 0 (bf16)
  for (int i = tid; i < HROWS * CINC; i += NTHREADS) {
    int r = i >> 7, c = i & 127;
    hl[r * HSTRIDE + c] = f2bf(x[(rowbase + r) * CINC + c]);
  }
  __syncthreads();

  // Aggregation: H_k[r,c] = sum_m T_k[n,m] * X[bl*17+m, c]  (k = 1,2)
  for (int i = tid; i < 2 * HROWS * CINC; i += NTHREADS) {
    int plane = 1 + (i >= HROWS * CINC);
    int j = (plane == 2) ? (i - HROWS * CINC) : i;
    int r = j >> 7, c = j & 127;
    int bl = r / NN;
    int n  = r - bl * NN;
    const float* Tk = tl + (plane - 1) * NN * NN + n * NN;
    const unsigned short* xcol = hl + (bl * NN) * HSTRIDE + c;
    float acc = 0.f;
#pragma unroll
    for (int m = 0; m < NN; ++m) acc += Tk[m] * bf2f(xcol[m * HSTRIDE]);
    hl[plane * (HROWS * HSTRIDE) + r * HSTRIDE + c] = f2bf(acc);
  }
  __syncthreads();

  // GEMM: each wave owns one 16-row tile (rt = wave, 0..16).
  // No row-tile loop => nothing for LICM to hoist => no spills.
  const int rt   = tid >> 5;           // wave id == row tile
  const int lane = tid & 31;
  const int lrow = lane & 15;
  const int lhi  = lane >> 4;

  v8f acc[8] = {};

#pragma unroll 1
  for (int k = 0; k < KORD; ++k) {
#pragma unroll 1
    for (int kc = 0; kc < 4; ++kc) {
      // A fragment: lanes 0-15 -> K {0..7,16..23}; lanes 16-31 -> K {8..15,24..31}
      const unsigned short* arow = hl + k * (HROWS * HSTRIDE)
          + (rt * 16 + lrow) * HSTRIDE + kc * 32 + lhi * 8;
      B32x2 araw;
      araw.lo = *(const uint4*)(arow);
      araw.hi = *(const uint4*)(arow + 16);
      v16bf A = __builtin_bit_cast(v16bf, araw);
#pragma unroll
      for (int ct = 0; ct < 8; ++ct) {
        const unsigned int* fb = wl + (((k * 4 + kc) * 8) + ct) * 256;
        B32x2 braw;
        braw.lo = *(const uint4*)(fb + lane * 4);
        braw.hi = *(const uint4*)(fb + 128 + lane * 4);
        v16bf Bm = __builtin_bit_cast(v16bf, braw);
        acc[ct] = __builtin_amdgcn_wmma_f32_16x16x32_bf16(
            false, A, false, Bm, (short)0, acc[ct], false, false);
      }
    }
  }

  // Epilogue: C/D layout -> VGPR r = row r (+8 for lanes 16-31), lane&15 = col
#pragma unroll
  for (int ct = 0; ct < 8; ++ct) {
    int col = ct * 16 + lrow;
    float bv = bias[col];
#pragma unroll
    for (int r8 = 0; r8 < 8; ++r8) {
      int row = rt * 16 + r8 + lhi * 8;
      float v = acc[ct][r8] + bv;
      v = v > 0.f ? v : 0.f;
      out[(rowbase + row) * COUTC + col] = v;
    }
  }
}

extern "C" void kernel_launch(void* const* d_in, const int* in_sizes, int n_in,
                              void* d_out, int out_size, void* d_ws, size_t ws_size,
                              hipStream_t stream) {
  const float* x      = (const float*)d_in[0];
  const float* adj    = (const float*)d_in[1];
  const float* weight = (const float*)d_in[2];
  const float* bias   = (const float*)d_in[3];

  unsigned int* wpack = (unsigned int*)d_ws;                    // 98304 B
  float*        Tmat  = (float*)((char*)d_ws + LDS_W_BYTES);    // 2312 B

  cheb_setup<<<1, 256, 0, stream>>>(adj, weight, Tmat, wpack);
  cheb_main<<<BQ / BB, NTHREADS, LDS_TOTAL, stream>>>(
      x, bias, Tmat, wpack, (float*)d_out);
}
